// SwiGLUExpert_32822140076135
// MI455X (gfx1250) — matrix operand order repaired and checked by reading, not compiled, not run on hardware
//
#include <hip/hip_runtime.h>

// ---------------------------------------------------------------------------
// SwiGLU MoE expert FFN for gfx1250 (MI455X).
//   E=8, C=2048, D=1024, I=2816
// bf16 WMMA (v_wmma_f32_16x16x32_bf16), fp32 accumulate,
// double-buffered LDS tiles filled with GLOBAL_LOAD_ASYNC_TO_LDS_B128
// (ASYNCcnt-tracked DMA) so tile fill overlaps the WMMA stream.
// All fragments use distinct registers so ds_load latency hides under WMMAs.
// ---------------------------------------------------------------------------

typedef unsigned short u16;
typedef unsigned int   u32;

typedef __bf16 bf16_t;
typedef bf16_t v16bf __attribute__((ext_vector_type(16)));
typedef float  v8f   __attribute__((ext_vector_type(8)));

#define E_DIM 8
#define C_DIM 2048
#define D_DIM 1024
#define I_DIM 2816

struct alignas(16) U128 { u32 x, y, z, w; };
struct alignas(16) F128 { float x, y, z, w; };
struct alignas(8)  US4  { u16 a, b, c, d; };

union Frag { U128 q[2]; v16bf v; };

#define WMMA_BF16(a, b, c) \
  __builtin_amdgcn_wmma_f32_16x16x32_bf16(false, (a).v, false, (b).v, \
                                          (short)0, (c), false, false)

// Async global -> LDS DMA (gfx1250). INST_OFFSET is added to both the LDS
// address (VDST vgpr) and the global address (VADDR vgpr pair), so one base
// pair covers a contiguous per-thread segment.
#define ASYNC_LD_B128(ldsoff, gptr, OFF)                                   \
  asm volatile("global_load_async_to_lds_b128 %0, %1, off offset:" OFF     \
               :: "v"(ldsoff), "v"(gptr) : "memory")

#define WAIT_ASYNC(N) asm volatile("s_wait_asynccnt " N ::: "memory")

__device__ __forceinline__ u16 f2bf(float f) {
  u32 u = __float_as_uint(f);
  u32 r = u + 0x7FFFu + ((u >> 16) & 1u);
  return (u16)(r >> 16);
}

__device__ __forceinline__ float silu(float g) {
  return g / (1.0f + __expf(-g));
}

__device__ __forceinline__ u32 lds_off(const void* p) {
  // generic pointer to LDS: low 32 bits are the LDS byte address (ISA 10.2)
  return (u32)(size_t)p;
}

// A fragment (16x32 bf16, ISA 7.12.2): lanes 0-15 hold K{0..7,16..23},
// lanes 16-31 hold K{8..15,24..31}; tile row stride = 64 elements.
__device__ __forceinline__ Frag loadA(const u16* s, int row, int kk, int half) {
  Frag f;
  f.q[0] = *(const U128*)&s[row * 64 + kk + half * 8];
  f.q[1] = *(const U128*)&s[row * 64 + kk + 16 + half * 8];
  return f;
}

// B fragment (32x16 bf16): lanes 0-15 hold K{0..7,16..23}, lanes 16-31 K{8..15,24..31};
// tile stored [n][k], row stride 64 elements.
__device__ __forceinline__ Frag loadB(const u16* s, int row, int kk, int half) {
  Frag f;
  f.q[0] = *(const U128*)&s[row * 64 + kk + half * 8];
  f.q[1] = *(const U128*)&s[row * 64 + kk + 16 + half * 8];
  return f;
}

// ---------------------------------------------------------------------------
// Stage 0a: elementwise fp32 -> bf16 (4 per thread)
// ---------------------------------------------------------------------------
__global__ __launch_bounds__(256) void cvt_bf16_kernel(const float* __restrict__ in,
                                                       u16* __restrict__ out) {
  size_t i4 = (size_t)blockIdx.x * blockDim.x + threadIdx.x;
  F128 f = ((const F128*)in)[i4];
  US4 o;
  o.a = f2bf(f.x); o.b = f2bf(f.y); o.c = f2bf(f.z); o.d = f2bf(f.w);
  ((US4*)out)[i4] = o;
}

// ---------------------------------------------------------------------------
// Stage 0b: per-expert transpose+convert: [E][R][Cc] fp32 -> [E][Cc][R] bf16
// ---------------------------------------------------------------------------
__global__ __launch_bounds__(256) void cvt_T_kernel(const float* __restrict__ in,
                                                    u16* __restrict__ out,
                                                    int R, int Cc) {
  size_t base = (size_t)blockIdx.z * (size_t)R * (size_t)Cc;
  size_t idx  = (size_t)blockIdx.x * 256 + threadIdx.x;
  int r = (int)(idx / (size_t)Cc);
  int c = (int)(idx % (size_t)Cc);
  out[base + (size_t)c * R + r] = f2bf(in[base + idx]);
}

// ---------------------------------------------------------------------------
// Stage 1: hidden = silu(x@Wg) * (x@Wu).
//   Block tile 128(C) x 64(I), BK=64, 8 waves; wave owns 32x32 of gate AND up
//   -> 16 WMMAs per K-step per wave. Double-buffered async LDS fill.
// ---------------------------------------------------------------------------
__global__ __launch_bounds__(256) void gateup_wmma(
    const u16* __restrict__ xbf,   // [E][C][D] bf16
    const u16* __restrict__ wgT,   // [E][I][D] bf16 (transposed)
    const u16* __restrict__ wuT,   // [E][I][D] bf16 (transposed)
    u16* __restrict__ hid)         // [E][C][I] bf16
{
  const int e  = blockIdx.z;
  const int cT = blockIdx.x * 128;
  const int iT = blockIdx.y * 64;
  const u16* X  = xbf + (size_t)e * C_DIM * D_DIM;
  const u16* Gm = wgT + (size_t)e * (size_t)I_DIM * D_DIM;
  const u16* Um = wuT + (size_t)e * (size_t)I_DIM * D_DIM;
  u16*       H  = hid + (size_t)e * (size_t)C_DIM * I_DIM;

  __shared__ alignas(16) u16 As[2][128 * 64];  // 2 x 16 KB
  __shared__ alignas(16) u16 Gs[2][64 * 64];   // 2 x  8 KB
  __shared__ alignas(16) u16 Us[2][64 * 64];   // 2 x  8 KB

  const int tid  = threadIdx.x;
  const int wave = tid >> 5;
  const int lane = tid & 31;
  const int half = lane >> 4;
  const int l    = lane & 15;
  const int m0   = (wave & 3) * 32;   // 4 M-groups of 32 rows
  const int n0   = (wave >> 2) * 32;  // 2 N-groups of 32 cols

  // async-fill mapping: A: 64B/thread of a 128x64 tile; G/U: 32B/thread of 64x64
  const int rowA = tid >> 1, colA = (tid & 1) * 32;
  const int rowG = tid >> 2, colG = (tid & 3) * 16;

  const u32 laBase = lds_off(&As[0][rowA * 64 + colA]);
  const u32 lgBase = lds_off(&Gs[0][rowG * 64 + colG]);
  const u32 luBase = lds_off(&Us[0][rowG * 64 + colG]);

  const u16* gpA = X  + (size_t)(cT + rowA) * D_DIM + colA;
  const u16* gpG = Gm + (size_t)(iT + rowG) * D_DIM + colG;
  const u16* gpU = Um + (size_t)(iT + rowG) * D_DIM + colG;

  auto issue = [&](int b) {
    const u32 la = laBase + (u32)b * (u32)sizeof(As[0]);
    const u32 lg = lgBase + (u32)b * (u32)sizeof(Gs[0]);
    const u32 lu = luBase + (u32)b * (u32)sizeof(Us[0]);
    ASYNC_LD_B128(la, gpA, "0");  ASYNC_LD_B128(la, gpA, "16");
    ASYNC_LD_B128(la, gpA, "32"); ASYNC_LD_B128(la, gpA, "48");
    ASYNC_LD_B128(lg, gpG, "0");  ASYNC_LD_B128(lg, gpG, "16");
    ASYNC_LD_B128(lu, gpU, "0");  ASYNC_LD_B128(lu, gpU, "16");
    gpA += 64; gpG += 64; gpU += 64;
  };

  v8f aG00 = {}, aG01 = {}, aG10 = {}, aG11 = {};
  v8f aU00 = {}, aU01 = {}, aU10 = {}, aU11 = {};

  const int NSTEP = D_DIM / 64;  // 16
  issue(0);

  for (int s = 0; s < NSTEP; ++s) {
    const int cur = s & 1;
    if (s + 1 < NSTEP) {
      issue(cur ^ 1);
      WAIT_ASYNC("8");   // current tile's 8 DMAs drained; next 8 in flight
    } else {
      WAIT_ASYNC("0");
    }
    __syncthreads();

    const u16* sA = &As[cur][0];
    const u16* sG = &Gs[cur][0];
    const u16* sU = &Us[cur][0];

#pragma unroll
    for (int kk = 0; kk < 64; kk += 32) {
      // Issue every fragment load up front (distinct registers) so the
      // scheduler can stagger s_wait_dscnt and overlap DS with WMMA.
      Frag a0  = loadA(sA, m0 + l,      kk, half);
      Frag a1  = loadA(sA, m0 + 16 + l, kk, half);
      Frag bg0 = loadB(sG, n0 + l,      kk, half);
      Frag bg1 = loadB(sG, n0 + 16 + l, kk, half);
      Frag bu0 = loadB(sU, n0 + l,      kk, half);
      Frag bu1 = loadB(sU, n0 + 16 + l, kk, half);

      aG00 = WMMA_BF16(a0, bg0, aG00);
      aG10 = WMMA_BF16(a1, bg0, aG10);
      aG01 = WMMA_BF16(a0, bg1, aG01);
      aG11 = WMMA_BF16(a1, bg1, aG11);
      aU00 = WMMA_BF16(a0, bu0, aU00);
      aU10 = WMMA_BF16(a1, bu0, aU10);
      aU01 = WMMA_BF16(a0, bu1, aU01);
      aU11 = WMMA_BF16(a1, bu1, aU11);
    }
    __syncthreads();  // done reading buf[cur] before it is refilled
  }

  // C/D layout: VGPR r -> row = half*8 + r, col = lane&15
#pragma unroll
  for (int r = 0; r < 8; ++r) {
    const size_t row0 = (size_t)(cT + m0 + half * 8 + r);
    const size_t row1 = row0 + 16;
    const size_t colb = (size_t)(iT + n0 + l);
    H[row0 * I_DIM + colb]      = f2bf(silu(aG00[r]) * aU00[r]);
    H[row0 * I_DIM + colb + 16] = f2bf(silu(aG01[r]) * aU01[r]);
    H[row1 * I_DIM + colb]      = f2bf(silu(aG10[r]) * aU10[r]);
    H[row1 * I_DIM + colb + 16] = f2bf(silu(aG11[r]) * aU11[r]);
  }
}

// ---------------------------------------------------------------------------
// Stage 2: out = hidden @ Wd  (K = I = 2816), fp32 output.
//   Block tile 128(C) x 128(D), BK=64, 8 waves; wave owns 32x64
//   -> 16 WMMAs per K-step per wave.
// ---------------------------------------------------------------------------
__global__ __launch_bounds__(256) void down_wmma(
    const u16* __restrict__ hid,   // [E][C][I] bf16
    const u16* __restrict__ wdT,   // [E][D][I] bf16 (transposed)
    float* __restrict__ out)       // [E][C][D] fp32
{
  const int e  = blockIdx.z;
  const int cT = blockIdx.x * 128;
  const int dT = blockIdx.y * 128;
  const u16* Hp = hid + (size_t)e * (size_t)C_DIM * I_DIM;
  const u16* W  = wdT + (size_t)e * (size_t)D_DIM * I_DIM;
  float*     O  = out + (size_t)e * C_DIM * D_DIM;

  __shared__ alignas(16) u16 As[2][128 * 64];  // 2 x 16 KB
  __shared__ alignas(16) u16 Bs[2][128 * 64];  // 2 x 16 KB

  const int tid  = threadIdx.x;
  const int wave = tid >> 5;
  const int lane = tid & 31;
  const int half = lane >> 4;
  const int l    = lane & 15;
  const int m0   = (wave & 3) * 32;   // 4 M-groups of 32 rows
  const int n0   = (wave >> 2) * 64;  // 2 N-groups of 64 cols

  const int rowA = tid >> 1, colA = (tid & 1) * 32;  // 64B/thread per tile

  const u32 laBase = lds_off(&As[0][rowA * 64 + colA]);
  const u32 lbBase = lds_off(&Bs[0][rowA * 64 + colA]);

  const u16* gpA = Hp + (size_t)(cT + rowA) * I_DIM + colA;
  const u16* gpB = W  + (size_t)(dT + rowA) * I_DIM + colA;

  auto issue = [&](int b) {
    const u32 la = laBase + (u32)b * (u32)sizeof(As[0]);
    const u32 lb = lbBase + (u32)b * (u32)sizeof(Bs[0]);
    ASYNC_LD_B128(la, gpA, "0");  ASYNC_LD_B128(la, gpA, "16");
    ASYNC_LD_B128(la, gpA, "32"); ASYNC_LD_B128(la, gpA, "48");
    ASYNC_LD_B128(lb, gpB, "0");  ASYNC_LD_B128(lb, gpB, "16");
    ASYNC_LD_B128(lb, gpB, "32"); ASYNC_LD_B128(lb, gpB, "48");
    gpA += 64; gpB += 64;
  };

  v8f d00 = {}, d01 = {}, d02 = {}, d03 = {};
  v8f d10 = {}, d11 = {}, d12 = {}, d13 = {};

  const int NSTEP = I_DIM / 64;  // 44
  issue(0);

  for (int s = 0; s < NSTEP; ++s) {
    const int cur = s & 1;
    if (s + 1 < NSTEP) {
      issue(cur ^ 1);
      WAIT_ASYNC("8");
    } else {
      WAIT_ASYNC("0");
    }
    __syncthreads();

    const u16* sA = &As[cur][0];
    const u16* sB = &Bs[cur][0];

#pragma unroll
    for (int kk = 0; kk < 64; kk += 32) {
      Frag a0 = loadA(sA, m0 + l,      kk, half);
      Frag a1 = loadA(sA, m0 + 16 + l, kk, half);
      Frag b0 = loadB(sB, n0 + l,      kk, half);
      Frag b1 = loadB(sB, n0 + 16 + l, kk, half);
      Frag b2 = loadB(sB, n0 + 32 + l, kk, half);
      Frag b3 = loadB(sB, n0 + 48 + l, kk, half);

      d00 = WMMA_BF16(a0, b0, d00);
      d10 = WMMA_BF16(a1, b0, d10);
      d01 = WMMA_BF16(a0, b1, d01);
      d11 = WMMA_BF16(a1, b1, d11);
      d02 = WMMA_BF16(a0, b2, d02);
      d12 = WMMA_BF16(a1, b2, d12);
      d03 = WMMA_BF16(a0, b3, d03);
      d13 = WMMA_BF16(a1, b3, d13);
    }
    __syncthreads();
  }

#pragma unroll
  for (int r = 0; r < 8; ++r) {
    const size_t row0 = (size_t)(cT + m0 + half * 8 + r);
    const size_t row1 = row0 + 16;
    const size_t colb = (size_t)(dT + n0 + l);
    O[row0 * D_DIM + colb]      = d00[r];
    O[row0 * D_DIM + colb + 16] = d01[r];
    O[row0 * D_DIM + colb + 32] = d02[r];
    O[row0 * D_DIM + colb + 48] = d03[r];
    O[row1 * D_DIM + colb]      = d10[r];
    O[row1 * D_DIM + colb + 16] = d11[r];
    O[row1 * D_DIM + colb + 32] = d12[r];
    O[row1 * D_DIM + colb + 48] = d13[r];
  }
}

// ---------------------------------------------------------------------------
// Launcher. Workspace (bf16 elements):
//   xbf [E*C*D] | wgT [E*I*D] | wuT [E*I*D] | wdT [E*D*I] | hid [E*C*I]
// ---------------------------------------------------------------------------
extern "C" void kernel_launch(void* const* d_in, const int* in_sizes, int n_in,
                              void* d_out, int out_size, void* d_ws, size_t ws_size,
                              hipStream_t stream) {
  (void)in_sizes; (void)n_in; (void)out_size; (void)ws_size;

  const float* x  = (const float*)d_in[0];
  const float* wg = (const float*)d_in[1];
  const float* wu = (const float*)d_in[2];
  const float* wd = (const float*)d_in[3];
  float* out = (float*)d_out;

  const size_t XN = (size_t)E_DIM * C_DIM * D_DIM;
  const size_t WN = (size_t)E_DIM * D_DIM * I_DIM;

  u16* xbf = (u16*)d_ws;
  u16* wgT = xbf + XN;
  u16* wuT = wgT + WN;
  u16* wdT = wuT + WN;
  u16* hid = wdT + WN;

  cvt_bf16_kernel<<<dim3((unsigned)(XN / 4 / 256)), 256, 0, stream>>>(x, xbf);
  cvt_T_kernel<<<dim3((D_DIM * I_DIM) / 256, 1, E_DIM), 256, 0, stream>>>(
      wg, wgT, D_DIM, I_DIM);
  cvt_T_kernel<<<dim3((D_DIM * I_DIM) / 256, 1, E_DIM), 256, 0, stream>>>(
      wu, wuT, D_DIM, I_DIM);
  cvt_T_kernel<<<dim3((I_DIM * D_DIM) / 256, 1, E_DIM), 256, 0, stream>>>(
      wd, wdT, I_DIM, D_DIM);

  gateup_wmma<<<dim3(C_DIM / 128, I_DIM / 64, E_DIM), 256, 0, stream>>>(
      xbf, wgT, wuT, hid);
  down_wmma<<<dim3(C_DIM / 128, D_DIM / 128, E_DIM), 256, 0, stream>>>(
      hid, wdT, out);
}
